// Gnn_38998303048251
// MI455X (gfx1250) — matrix operand order repaired and checked by reading, not compiled, not run on hardware
//
#include <hip/hip_runtime.h>
#include <hip/hip_fp16.h>

typedef __attribute__((ext_vector_type(16))) _Float16 v16h;
typedef __attribute__((ext_vector_type(8)))  _Float16 v8h;
typedef __attribute__((ext_vector_type(4)))  _Float16 v4h;
typedef __attribute__((ext_vector_type(8)))  float    v8f;
typedef __attribute__((ext_vector_type(4)))  float    v4f;

#define Bk   8
#define Nk   256
#define NNFk 64
#define NEFk 32
#define Hk   128
#define NNOk 64
#define NEOk 32
#define INk  160   // 2*NNF + NEF

// ---------------------------------------------------------------------------
// WMMA fragment helpers, f16 16x16x32.  Two operand orientations:
//
//  "transposed" GEMM (layers 1,2,3e):  D = W^T x Act^T  => D[chan,row]
//     A = weight^T  ([chan][K] LDS layout), B = act^T (row-major [row][K])
//     D: lane r=lane&15 (row), VGPR v -> chan kg*8+v => 8 contiguous chans
//        -> one ds_store_b128 per tile after packed cvt.
//
//  "natural" GEMM (layer 3n, needs per-lane row sums): D = Act x W
// ---------------------------------------------------------------------------

__device__ __forceinline__ v16h ld_wA(const _Float16* wt, int K, int c0, int kb, int lane) {
  const int kg = lane >> 4, m = lane & 15;
  const _Float16* p = wt + (size_t)(c0 + m) * K + kb + kg * 8;
  v8h lo = *(const v8h*)(p);
  v8h hi = *(const v8h*)(p + 16);
  return __builtin_shufflevector(lo, hi, 0,1,2,3,4,5,6,7,8,9,10,11,12,13,14,15);
}

__device__ __forceinline__ v16h ld_actB(const _Float16* act, int rs, int kb, int lane) {
  const int kg = lane >> 4, r = lane & 15;
  const _Float16* p = act + r * rs + kb + kg * 8;
  v8h lo = *(const v8h*)(p);
  v8h hi = *(const v8h*)(p + 16);
  return __builtin_shufflevector(lo, hi, 0,1,2,3,4,5,6,7,8,9,10,11,12,13,14,15);
}

__device__ __forceinline__ v16h ld_actA(const _Float16* act, int rs, int kb, int lane) {
  const int kg = lane >> 4, r = lane & 15;
  const _Float16* p = act + r * rs + kb + kg * 8;
  v8h lo = *(const v8h*)(p);
  v8h hi = *(const v8h*)(p + 16);
  return __builtin_shufflevector(lo, hi, 0,1,2,3,4,5,6,7,8,9,10,11,12,13,14,15);
}

__device__ __forceinline__ v16h ld_wB(const _Float16* wt, int K, int c0, int kb, int lane) {
  const int kg = lane >> 4, c = lane & 15;
  const _Float16* p = wt + (size_t)(c0 + c) * K + kb + kg * 8;
  v8h lo = *(const v8h*)(p);
  v8h hi = *(const v8h*)(p + 16);
  return __builtin_shufflevector(lo, hi, 0,1,2,3,4,5,6,7,8,9,10,11,12,13,14,15);
}

__device__ __forceinline__ v8f wmma16(v16h a, v16h b, v8f c) {
  return __builtin_amdgcn_wmma_f32_16x16x32_f16(false, a, false, b, (short)0, c, false, false);
}

// bias along D's M dim (chan) from LDS: per-lane 8 contiguous floats
__device__ __forceinline__ v8f bias8(const float* bias, int c0, int lane) {
  const int kg = lane >> 4;
  v4f b0 = *(const v4f*)(bias + c0 + kg * 8);
  v4f b1 = *(const v4f*)(bias + c0 + kg * 8 + 4);
  return __builtin_shufflevector(b0, b1, 0,1,2,3,4,5,6,7);
}

// relu as a single v_med3_f32: clamp(x, 0, +inf)
__device__ __forceinline__ float relu1(float x) {
  return __builtin_amdgcn_fmed3f(x, 0.f, __builtin_huge_valf());
}

__device__ __forceinline__ void st_cT_relu_f16(v8f acc, _Float16* dst, int rs, int c0, int lane) {
  const int kg = lane >> 4, r = lane & 15;
  v8h h;
#pragma unroll
  for (int v = 0; v < 8; ++v) h[v] = (_Float16)relu1(acc[v]);
  *(v8h*)(dst + r * rs + c0 + kg * 8) = h;
}

extern __shared__ _Float16 smem[];

__global__ __launch_bounds__(512, 1)
void Gnn_38998303048251_kernel(
    const float* __restrict__ node_feat, const float* __restrict__ edge_feat,
    const float* __restrict__ ew1, const float* __restrict__ eb1,
    const float* __restrict__ ew2, const float* __restrict__ eb2,
    const float* __restrict__ ew3, const float* __restrict__ eb3,
    const float* __restrict__ nw1, const float* __restrict__ nb1,
    const float* __restrict__ nw2, const float* __restrict__ nb2,
    const float* __restrict__ nw3, const float* __restrict__ nb3,
    const float* __restrict__ en_g, const float* __restrict__ en_b,
    const float* __restrict__ nn_g, const float* __restrict__ nn_b,
    float* __restrict__ node_out, float* __restrict__ edge_out)
{
  const int tid  = threadIdx.x;
  const int wave = tid >> 5;          // 0..15
  const int lane = tid & 31;
  const int wloc = wave & 7;          // wave within (b,i) group
  const int wgrp = wave >> 3;         // which (b,i) pair of this block

  const int bi = blockIdx.x * 2 + wgrp;
  const int b  = bi >> 8;
  const int i  = bi & 255;

  // ---- LDS carve ---------------------------------------------------------
  _Float16* sW1e = smem;                       // [128][160] = weight^T
  _Float16* sW1n = sW1e + Hk * INk;
  _Float16* sW2e = sW1n + Hk * INk;            // [128][128]
  _Float16* sW2n = sW2e + Hk * Hk;
  _Float16* sW3e = sW2n + Hk * Hk;             // [32][128]
  _Float16* sW3n = sW3e + NEOk * Hk;           // [64][128]
  _Float16* bufX = sW3n + NNOk * Hk;           // [16 waves][16 rows][256]
  float*    sNP  = (float*)(bufX + 16 * 16 * 256); // [32 partials][64]
  float*    sB1e = sNP + 32 * 64;              // biases (keep loads near use)
  float*    sB1n = sB1e + Hk;
  float*    sB2e = sB1n + Hk;
  float*    sB2n = sB2e + Hk;
  float*    sB3e = sB2n + Hk;                  // [32]
  float*    sB3n = sB3e + NEOk;                // [64]

  // ---- stage weights transposed -> f16 LDS (vectorized b128 stores) ------
  auto stage = [&](const float* __restrict__ w, _Float16* wt, int K, int NC) {
    const int K8 = K >> 3;
    for (int idx = tid; idx < NC * K8; idx += 512) {
      const int k8 = idx / NC, c = idx - k8 * NC;   // coalesced global reads
      v8h h;
#pragma unroll
      for (int u = 0; u < 8; ++u) h[u] = (_Float16)w[(k8 * 8 + u) * NC + c];
      *(v8h*)(wt + (size_t)c * K + k8 * 8) = h;
    }
  };
  stage(ew1, sW1e, INk, Hk);
  stage(nw1, sW1n, INk, Hk);
  stage(ew2, sW2e, Hk, Hk);
  stage(nw2, sW2n, Hk, Hk);
  stage(ew3, sW3e, Hk, NEOk);
  stage(nw3, sW3n, Hk, NNOk);
  if (tid < Hk) {
    sB1e[tid] = eb1[tid]; sB1n[tid] = nb1[tid];
    sB2e[tid] = eb2[tid]; sB2n[tid] = nb2[tid];
    if (tid < NEOk) sB3e[tid] = eb3[tid];
    if (tid < NNOk) sB3n[tid] = nb3[tid];
  }
  __syncthreads();

  const float* nodeB = node_feat + (size_t)b * Nk * NNFk;
  const float* nodeI = nodeB + (size_t)i * NNFk;

  _Float16* X = bufX + wave * 16 * 256;   // per-wave buffer: ef -> h1 -> h2

  for (int chunk = 0; chunk < 2; ++chunk) {
    const int j0 = chunk * 128 + wloc * 16;

    // ---- stage ef rows [node[j] | node[i] | edge[i,j]] as f16 ------------
    {
      const int r = lane >> 1, part = lane & 1;
      const int j = j0 + r;
      const float* nodeJ = nodeB + (size_t)j * NNFk;
      const float* edgeP = edge_feat + (((size_t)(b * Nk + i)) * Nk + j) * NEFk;
      if (chunk == 0) __builtin_prefetch(edgeP + 128 * NEFk, 0, 0);
#pragma unroll
      for (int q = 0; q < 20; ++q) {
        const int k = part * 80 + q * 4;
        const float* src = (k < NNFk)     ? (nodeJ + k)
                         : (k < 2 * NNFk) ? (nodeI + (k - NNFk))
                                          : (edgeP + (k - 2 * NNFk));
        const v4f vv = *(const v4f*)src;
        v4h h; h[0] = (_Float16)vv[0]; h[1] = (_Float16)vv[1];
               h[2] = (_Float16)vv[2]; h[3] = (_Float16)vv[3];
        *(v4h*)(X + r * 256 + k) = h;
      }
    }

    // ---- layer 1: h1 = relu(ef @ {ew1|nw1} + b1)  (K=160) ----------------
    {
      v16h Bf[5];                            // whole ef tile -> registers
#pragma unroll
      for (int kc = 0; kc < 5; ++kc) Bf[kc] = ld_actB(X, 256, kc * 32, lane);
#pragma unroll
      for (int half = 0; half < 2; ++half) {
        const _Float16* Wt = half ? sW1n : sW1e;
        const float* bias  = half ? sB1n : sB1e;
#pragma unroll
        for (int nt = 0; nt < 8; ++nt) {
          v16h W[5];
#pragma unroll
          for (int kc = 0; kc < 5; ++kc) W[kc] = ld_wA(Wt, INk, nt * 16, kc * 32, lane);
          v8f acc = bias8(bias, nt * 16, lane);
#pragma unroll
          for (int kc = 0; kc < 5; ++kc) acc = wmma16(W[kc], Bf[kc], acc);
          st_cT_relu_f16(acc, X, 256, half * 128 + nt * 16, lane);
        }
      }
    }

    // ---- layer 2: h2 = relu(h1 @ {ew2|nw2} + b2) -------------------------
    // each half reads AND writes the same 128-col range -> independent halves
#pragma unroll
    for (int half = 0; half < 2; ++half) {
      const _Float16* Wt = half ? sW2n : sW2e;
      const float* bias  = half ? sB2n : sB2e;
      v16h B2[4];
#pragma unroll
      for (int kc = 0; kc < 4; ++kc)
        B2[kc] = ld_actB(X, 256, half * 128 + kc * 32, lane);
#pragma unroll
      for (int nt = 0; nt < 8; ++nt) {
        v16h W[4];
#pragma unroll
        for (int kc = 0; kc < 4; ++kc) W[kc] = ld_wA(Wt, Hk, nt * 16, kc * 32, lane);
        v8f acc = bias8(bias, nt * 16, lane);
#pragma unroll
        for (int kc = 0; kc < 4; ++kc) acc = wmma16(W[kc], B2[kc], acc);
        st_cT_relu_f16(acc, X, 256, half * 128 + nt * 16, lane);
      }
    }

    // ---- layer 3 edge head: h2e (cols 0..127) -> LN -> edge_out ----------
    {
      v16h Bf3[4];
#pragma unroll
      for (int kc = 0; kc < 4; ++kc) Bf3[kc] = ld_actB(X, 256, kc * 32, lane);
      v8f accE[2];
#pragma unroll
      for (int nt = 0; nt < 2; ++nt) {
        v16h W[4];
#pragma unroll
        for (int kc = 0; kc < 4; ++kc) W[kc] = ld_wA(sW3e, Hk, nt * 16, kc * 32, lane);
        accE[nt] = bias8(sB3e, nt * 16, lane);
#pragma unroll
        for (int kc = 0; kc < 4; ++kc) accE[nt] = wmma16(W[kc], Bf3[kc], accE[nt]);
      }
      // lane holds 16 of this row's 32 channels; partner = lane^16
      float s = 0.f;
#pragma unroll
      for (int nt = 0; nt < 2; ++nt)
#pragma unroll
        for (int v = 0; v < 8; ++v) s += accE[nt][v];
      s += __shfl_xor(s, 16, 32);
      const float m = s * (1.f / 32.f);
      float vs = 0.f;
#pragma unroll
      for (int nt = 0; nt < 2; ++nt)
#pragma unroll
        for (int v = 0; v < 8; ++v) { float d = accE[nt][v] - m; vs += d * d; }
      vs += __shfl_xor(vs, 16, 32);
      const float inv = rsqrtf(vs * (1.f / 32.f) + 1e-5f);

      const int kg = lane >> 4, r = lane & 15;
      const int j = j0 + r;
      float* outp = edge_out + (((size_t)(b * Nk + i)) * Nk + j) * NEOk;
#pragma unroll
      for (int nt = 0; nt < 2; ++nt) {
        const int cb = nt * 16 + kg * 8;
        v4f g0 = *(const v4f*)(en_g + cb), g1 = *(const v4f*)(en_g + cb + 4);
        v4f o0 = *(const v4f*)(en_b + cb), o1 = *(const v4f*)(en_b + cb + 4);
#pragma unroll
        for (int u = 0; u < 4; ++u) {
          o0[u] += g0[u] * (accE[nt][u]     - m) * inv;
          o1[u] += g1[u] * (accE[nt][4 + u] - m) * inv;
        }
        *(v4f*)(outp + cb)     = o0;
        *(v4f*)(outp + cb + 4) = o1;
      }
    }

    // ---- layer 3 node head: h2n (cols 128..255), natural orientation ------
    {
      v16h A4[4];
#pragma unroll
      for (int kc = 0; kc < 4; ++kc) A4[kc] = ld_actA(X, 256, 128 + kc * 32, lane);
#pragma unroll
      for (int nt = 0; nt < 4; ++nt) {
        v16h W[4];
#pragma unroll
        for (int kc = 0; kc < 4; ++kc) W[kc] = ld_wB(sW3n, Hk, nt * 16, kc * 32, lane);
        const float bv = sB3n[nt * 16 + (lane & 15)];
        v8f acc = {bv, bv, bv, bv, bv, bv, bv, bv};
#pragma unroll
        for (int kc = 0; kc < 4; ++kc) acc = wmma16(A4[kc], W[kc], acc);
        const float sum = acc[0] + acc[1] + acc[2] + acc[3]
                        + acc[4] + acc[5] + acc[6] + acc[7];
        const int kg = lane >> 4;
        float* slot = &sNP[(wave * 2 + kg) * 64 + nt * 16 + (lane & 15)];
        if (chunk == 0) *slot = sum; else *slot += sum;   // unique writer
      }
    }
  }

  __syncthreads();

  // ---- node LayerNorm over 64 (deterministic), one wave per (b,i) --------
  if (wloc == 0) {
    const int base = wgrp * 16;             // this pair's 16 partial rows
    float v0 = 0.f, v1 = 0.f;
#pragma unroll
    for (int p = 0; p < 16; ++p) {
      v0 += sNP[(base + p) * 64 + lane];
      v1 += sNP[(base + p) * 64 + lane + 32];
    }
    float s = v0 + v1;
#pragma unroll
    for (int off = 16; off > 0; off >>= 1) s += __shfl_xor(s, off, 32);
    const float m = s * (1.f / 64.f);
    const float d0 = v0 - m, d1 = v1 - m;
    float vs = d0 * d0 + d1 * d1;
#pragma unroll
    for (int off = 16; off > 0; off >>= 1) vs += __shfl_xor(vs, off, 32);
    const float inv = rsqrtf(vs * (1.f / 64.f) + 1e-5f);
    float* outp = node_out + ((size_t)(b * Nk) + i) * NNOk;
    outp[lane]      = nn_g[lane]      * d0 * inv + nn_b[lane];
    outp[lane + 32] = nn_g[lane + 32] * d1 * inv + nn_b[lane + 32];
  }
}

extern "C" void kernel_launch(void* const* d_in, const int* in_sizes, int n_in,
                              void* d_out, int out_size, void* d_ws, size_t ws_size,
                              hipStream_t stream) {
  const float* node_feat = (const float*)d_in[0];
  const float* edge_feat = (const float*)d_in[1];
  const float* ew1 = (const float*)d_in[2];
  const float* eb1 = (const float*)d_in[3];
  const float* ew2 = (const float*)d_in[4];
  const float* eb2 = (const float*)d_in[5];
  const float* ew3 = (const float*)d_in[6];
  const float* eb3 = (const float*)d_in[7];
  const float* nw1 = (const float*)d_in[8];
  const float* nb1 = (const float*)d_in[9];
  const float* nw2 = (const float*)d_in[10];
  const float* nb2 = (const float*)d_in[11];
  const float* nw3 = (const float*)d_in[12];
  const float* nb3 = (const float*)d_in[13];
  const float* en_g = (const float*)d_in[14];
  const float* en_b = (const float*)d_in[15];
  const float* nn_g = (const float*)d_in[16];
  const float* nn_b = (const float*)d_in[17];

  float* node_out = (float*)d_out;                     // [8,256,64]
  float* edge_out = node_out + (size_t)Bk * Nk * NNOk; // [8,256,256,32]

  // weights 172032B + activations 131072B + partials 8192B + biases 2432B
  const size_t smemBytes =
      (size_t)(2 * Hk * INk + 2 * Hk * Hk + NEOk * Hk + NNOk * Hk
               + 16 * 16 * 256) * sizeof(_Float16)
      + (32 * 64 + 4 * Hk + NEOk + NNOk) * sizeof(float);

  (void)hipFuncSetAttribute((const void*)Gnn_38998303048251_kernel,
                            hipFuncAttributeMaxDynamicSharedMemorySize,
                            (int)smemBytes);

  Gnn_38998303048251_kernel<<<dim3(Bk * Nk / 2), dim3(512), smemBytes, stream>>>(
      node_feat, edge_feat, ew1, eb1, ew2, eb2, ew3, eb3,
      nw1, nb1, nw2, nb2, nw3, nb3, en_g, en_b, nn_g, nn_b,
      node_out, edge_out);
}
